// OffsetAttention_61899068670757
// MI455X (gfx1250) — matrix operand order repaired and checked by reading, not compiled, not run on hardware
//
#include <hip/hip_runtime.h>
#include <math.h>

typedef __bf16 bf16;
typedef bf16  v16bf __attribute__((ext_vector_type(16)));
typedef float v8f   __attribute__((ext_vector_type(8)));

constexpr int BB = 8;      // batch
constexpr int CC = 256;    // channels
constexpr int NN = 4096;   // points
constexpr int DD = 64;     // q/k dim

// ---------------------------------------------------------------- WMMA helpers
__device__ __forceinline__ v8f wmma_bf16(v16bf a, v16bf b, v8f c) {
  // v_wmma_f32_16x16x32_bf16: D = A(16x32) * B(32x16) + C, f32 accum
  return __builtin_amdgcn_wmma_f32_16x16x32_bf16(false, a, false, b, (short)0, c,
                                                 false, false);
}

// A-fragment (16-bit, 16x32): lane = row (M = lane&15), h = lane>>4.
// Per-lane K values: {16h+0..15}.  p must point at rowBase + 16h.
__device__ __forceinline__ v16bf loadA16(const bf16* p) {
  union { uint4 u[2]; v16bf v; } r;
  r.u[0] = *reinterpret_cast<const uint4*>(p);
  r.u[1] = *reinterpret_cast<const uint4*>(p + 8);
  return r.v;
}
// B-fragment (16-bit, 32x16): lane = col (N = lane&15), h = lane>>4.
// Per-lane K values: {16h+0..15}.  p must point at colBase + 16h (K-contig).
__device__ __forceinline__ v16bf loadB16(const bf16* p) {
  union { uint4 u[2]; v16bf v; } r;
  r.u[0] = *reinterpret_cast<const uint4*>(p);
  r.u[1] = *reinterpret_cast<const uint4*>(p + 8);
  return r.v;
}

// ---------------------------------------------------------------- prep kernels
// x[B][C][N] f32 -> xT[B][N][C] bf16 (LDS tile transpose, coalesced both sides)
__global__ __launch_bounds__(256) void k_xpose(const float* __restrict__ x,
                                               bf16* __restrict__ xT) {
  __shared__ float t[16][17];
  const int b = blockIdx.z, c0 = blockIdx.y * 16, n0 = blockIdx.x * 16;
  const float* xb = x + (size_t)b * CC * NN;
  t[threadIdx.y][threadIdx.x] =
      xb[(size_t)(c0 + threadIdx.y) * NN + n0 + threadIdx.x];
  __syncthreads();
  xT[((size_t)b * NN + n0 + threadIdx.y) * CC + c0 + threadIdx.x] =
      (bf16)t[threadIdx.x][threadIdx.y];
}

__global__ __launch_bounds__(256) void k_cvt(bf16* __restrict__ dst,
                                             const float* __restrict__ src, int n) {
  int i = blockIdx.x * blockDim.x + threadIdx.x;
  if (i < n) dst[i] = (bf16)src[i];
}

__global__ __launch_bounds__(256) void k_bnprep(const float* __restrict__ g,
                                                const float* __restrict__ bt,
                                                const float* __restrict__ mn,
                                                const float* __restrict__ vr,
                                                float* __restrict__ inv,
                                                float* __restrict__ ofs) {
  int i = threadIdx.x;
  if (i < CC) {
    float iv = g[i] * rsqrtf(vr[i] + 1e-5f);
    inv[i] = iv;
    ofs[i] = bt[i] - mn[i] * iv;
  }
}

// ---------------------------------------------------------------- projections
// O[b][n][d] = sum_c xT[b][n][c] * W[d][c]   (row-major output [N][Dout])
__global__ __launch_bounds__(256) void k_proj(const bf16* __restrict__ xT,
                                              const bf16* __restrict__ W,
                                              bf16* __restrict__ O, int Dout) {
  const int lane = threadIdx.x & 31, l = lane & 15, h = lane >> 4;
  const int gw = blockIdx.x * (blockDim.x >> 5) + (threadIdx.x >> 5);
  const int tilesD = Dout >> 4, tilesPerB = (NN >> 4) * tilesD;
  const int b = gw / tilesPerB, t = gw % tilesPerB;
  const int n0 = (t / tilesD) << 4, d0 = (t % tilesD) << 4;

  const bf16* arow = xT + ((size_t)b * NN + n0 + l) * CC + 16 * h;
  const bf16* brow = W + (size_t)(d0 + l) * CC + 16 * h;
  v8f acc = {};
#pragma unroll
  for (int k = 0; k < CC; k += 32)
    acc = wmma_bf16(loadA16(arow + k), loadB16(brow + k), acc);

  bf16* ob = O + (size_t)b * NN * Dout;
#pragma unroll
  for (int r = 0; r < 8; ++r)
    ob[(size_t)(n0 + r + 8 * h) * Dout + d0 + l] = (bf16)acc[r];
}

// OT[b][d][n] = sum_c xT[b][n][c] * W[d][c]  (TRANSPOSED output [Dout][N]).
// D-fragment: lane holds 8 consecutive rows n for fixed col d -> one b128 store.
__global__ __launch_bounds__(256) void k_projT(const bf16* __restrict__ xT,
                                               const bf16* __restrict__ W,
                                               bf16* __restrict__ OT, int Dout) {
  const int lane = threadIdx.x & 31, l = lane & 15, h = lane >> 4;
  const int gw = blockIdx.x * (blockDim.x >> 5) + (threadIdx.x >> 5);
  const int tilesD = Dout >> 4, tilesPerB = (NN >> 4) * tilesD;
  const int b = gw / tilesPerB, t = gw % tilesPerB;
  const int n0 = (t / tilesD) << 4, d0 = (t % tilesD) << 4;

  const bf16* arow = xT + ((size_t)b * NN + n0 + l) * CC + 16 * h;
  const bf16* brow = W + (size_t)(d0 + l) * CC + 16 * h;
  v8f acc = {};
#pragma unroll
  for (int k = 0; k < CC; k += 32)
    acc = wmma_bf16(loadA16(arow + k), loadB16(brow + k), acc);

  union { uint4 u; bf16 e[8]; } st;
#pragma unroll
  for (int r = 0; r < 8; ++r) st.e[r] = (bf16)acc[r];
  // n = n0 + 8h + r contiguous over r; 16B aligned (NN*2, n0*2, 16h bytes)
  *reinterpret_cast<uint4*>(OT + ((size_t)b * Dout + d0 + l) * NN + n0 + 8 * h) =
      st.u;
}

// ---------------------------------------------------------------- pass 1
// Column softmax stats over the query axis n: colmax[m], colsum[m] (online).
__global__ __launch_bounds__(256) void k_colstats(const bf16* __restrict__ q,
                                                  const bf16* __restrict__ k,
                                                  float* __restrict__ colmax,
                                                  float* __restrict__ colsum) {
  const int lane = threadIdx.x & 31, l = lane & 15, h = lane >> 4;
  const int gw = blockIdx.x * (blockDim.x >> 5) + (threadIdx.x >> 5);
  const int b = gw >> 8, m0 = (gw & 255) << 4;

  const bf16* qb = q + (size_t)b * NN * DD;
  const bf16* kp = k + ((size_t)b * NN + m0 + l) * DD + 16 * h;
  const v16bf bk0 = loadB16(kp);        // d = 0..31
  const v16bf bk1 = loadB16(kp + 32);   // d = 32..63

  float rmax = -INFINITY, rsum = 0.f;
  for (int n0 = 0; n0 < NN; n0 += 16) {
    const bf16* qrow = qb + (size_t)(n0 + l) * DD + 16 * h;
    __builtin_prefetch(qrow + 16 * DD, 0, 0);
    v8f e = {};
    e = wmma_bf16(loadA16(qrow), bk0, e);
    e = wmma_bf16(loadA16(qrow + 32), bk1, e);
    float lm = e[0];
#pragma unroll
    for (int r = 1; r < 8; ++r) lm = fmaxf(lm, e[r]);
    lm = fmaxf(lm, __shfl_xor(lm, 16, 32));   // combine rows 0..7 / 8..15
    const float nm = fmaxf(rmax, lm);
    const float sc = __expf(rmax - nm);       // exp(-inf)=0 on first tile
    float ps = 0.f;
#pragma unroll
    for (int r = 0; r < 8; ++r) ps += __expf(e[r] - nm);
    ps += __shfl_xor(ps, 16, 32);
    rsum = rsum * sc + ps;
    rmax = nm;
  }
  if (h == 0) {
    colmax[(size_t)b * NN + m0 + l] = rmax;
    colsum[(size_t)b * NN + m0 + l] = rsum;
  }
}

// ---------------------------------------------------------------- pass 2
// Per 16-row strip: recompute E, w = exp(E-colmax)/colsum, acc += w*V,
// rowsum += sum_m w; finally y = x - acc/rowsum, stored bf16 [N][C].
// vT is [B][C][N]: the B-fragment K axis (m) is contiguous -> b128 loads.
__global__ __launch_bounds__(128) void k_fsa(const bf16* __restrict__ q,
                                             const bf16* __restrict__ k,
                                             const bf16* __restrict__ vT,
                                             const float* __restrict__ colmax,
                                             const float* __restrict__ colsum,
                                             const float* __restrict__ x,
                                             bf16* __restrict__ y) {
  __shared__ bf16 wlds[16][136];   // 16 rows x 128 cols, padded: row = 272B (16B-aligned)
  __shared__ float rowsum[16];

  const int lane = threadIdx.x & 31, l = lane & 15, h = lane >> 4;
  const int wv = threadIdx.x >> 5;            // wave 0..3
  const int b = blockIdx.x >> 8, n0 = (blockIdx.x & 255) << 4;

  const bf16* qrow = q + ((size_t)b * NN + n0 + l) * DD + 16 * h;
  const v16bf aq0 = loadA16(qrow);
  const v16bf aq1 = loadA16(qrow + 32);
  const bf16* kb = k + (size_t)b * NN * DD;
  const bf16* vb = vT + (size_t)b * CC * NN;
  const float* cmx = colmax + (size_t)b * NN;
  const float* csm = colsum + (size_t)b * NN;

  v8f acc[4] = {};
  if (threadIdx.x < 16) rowsum[threadIdx.x] = 0.f;
  __syncthreads();

  for (int mb = 0; mb < NN; mb += 128) {
    // --- my 32 columns: two 16x16 E tiles -> w into LDS
#pragma unroll
    for (int t = 0; t < 2; ++t) {
      const int wcol = wv * 32 + t * 16 + l;
      const int mcol = mb + wcol;
      const bf16* kp = kb + (size_t)mcol * DD + 16 * h;
      v8f e = {};
      e = wmma_bf16(aq0, loadB16(kp), e);
      e = wmma_bf16(aq1, loadB16(kp + 32), e);
      const float cm = cmx[mcol];
      const float ci = 1.0f / csm[mcol];
#pragma unroll
      for (int r = 0; r < 8; ++r)
        wlds[r + 8 * h][wcol] = (bf16)(__expf(e[r] - cm) * ci);
    }
    __syncthreads();

    // --- rowsum partial: wave wv owns rows 4*wv..4*wv+3
#pragma unroll
    for (int rr = 0; rr < 4; ++rr) {
      const int row = wv * 4 + rr;
      float s = (float)wlds[row][lane] + (float)wlds[row][lane + 32] +
                (float)wlds[row][lane + 64] + (float)wlds[row][lane + 96];
#pragma unroll
      for (int o = 16; o; o >>= 1) s += __shfl_xor(s, o, 32);
      if (lane == 0) rowsum[row] += s;
    }

    // --- acc += w(16x128) * V(128 x myC64): 4 K-chunks x 4 C-tiles
#pragma unroll
    for (int kc = 0; kc < 4; ++kc) {
      const v16bf aw = loadA16(&wlds[l][kc * 32 + 16 * h]);
#pragma unroll
      for (int ct = 0; ct < 4; ++ct) {
        const int c = wv * 64 + ct * 16 + l;
        const bf16* vp = vb + (size_t)c * NN + mb + kc * 32 + 16 * h;
        acc[ct] = wmma_bf16(aw, loadB16(vp), acc[ct]);
      }
    }
    __syncthreads();
  }

  // --- epilogue: fsa = acc/rowsum ; y = x - fsa  (bf16, [N][C] row-major)
#pragma unroll
  for (int ct = 0; ct < 4; ++ct) {
    const int c = wv * 64 + ct * 16 + l;
#pragma unroll
    for (int r = 0; r < 8; ++r) {
      const int nl = r + 8 * h, n = n0 + nl;
      const float fsa = acc[ct][r] / (1e-9f + rowsum[nl]);
      const float yv = x[((size_t)b * CC + c) * NN + n] - fsa;
      y[((size_t)b * NN + n) * CC + c] = (bf16)yv;
    }
  }
}

// ---------------------------------------------------------------- final GEMM
// out[b][d][n] = relu( (Wp . y)[d][n] * bninv[d] + bnofs[d] ) + x[b][d][n]
__global__ __launch_bounds__(256) void k_final(const bf16* __restrict__ y,
                                               const bf16* __restrict__ Wp,
                                               const float* __restrict__ bninv,
                                               const float* __restrict__ bnofs,
                                               const float* __restrict__ x,
                                               float* __restrict__ out) {
  const int lane = threadIdx.x & 31, l = lane & 15, h = lane >> 4;
  const int gw = blockIdx.x * (blockDim.x >> 5) + (threadIdx.x >> 5);
  const int tilesPerB = (NN >> 4) * (CC >> 4);
  const int b = gw / tilesPerB, t = gw % tilesPerB;
  const int n0 = (t / (CC >> 4)) << 4, d0 = (t % (CC >> 4)) << 4;

  const bf16* arow = y + ((size_t)b * NN + n0 + l) * CC + 16 * h;
  const bf16* brow = Wp + (size_t)(d0 + l) * CC + 16 * h;
  v8f acc = {};
#pragma unroll
  for (int kk = 0; kk < CC; kk += 32)
    acc = wmma_bf16(loadA16(arow + kk), loadB16(brow + kk), acc);

  const int d = d0 + l;
  const float iv = bninv[d], of = bnofs[d];
  const float* xr = x + ((size_t)b * CC + d) * NN;
  float* orow = out + ((size_t)b * CC + d) * NN;
#pragma unroll
  for (int r = 0; r < 8; ++r) {
    const int n = n0 + r + 8 * h;
    const float hv = fmaxf(acc[r] * iv + of, 0.0f);
    orow[n] = hv + xr[n];
  }
}

// ---------------------------------------------------------------- launcher
extern "C" void kernel_launch(void* const* d_in, const int* in_sizes, int n_in,
                              void* d_out, int out_size, void* d_ws, size_t ws_size,
                              hipStream_t stream) {
  const float* x   = (const float*)d_in[0];
  const float* Wq  = (const float*)d_in[1];
  const float* Wk  = (const float*)d_in[2];
  const float* Wv  = (const float*)d_in[3];
  const float* Wp  = (const float*)d_in[4];
  const float* bg  = (const float*)d_in[5];
  const float* bb  = (const float*)d_in[6];
  const float* bm  = (const float*)d_in[7];
  const float* bv  = (const float*)d_in[8];
  float* out = (float*)d_out;

  char* ws = (char*)d_ws;
  size_t off = 0;
  auto take = [&](size_t bytes) -> void* {
    void* p = ws + off;
    off += (bytes + 255) & ~(size_t)255;
    return p;
  };
  bf16* xT  = (bf16*)take((size_t)BB * NN * CC * 2);   // 16 MB  [B][N][C]
  bf16* qb  = (bf16*)take((size_t)BB * NN * DD * 2);   // 4 MB   [B][N][64]
  bf16* kb  = (bf16*)take((size_t)BB * NN * DD * 2);   // 4 MB   [B][N][64]
  bf16* vTb = (bf16*)take((size_t)BB * CC * NN * 2);   // 16 MB  [B][C][N]
  bf16* yb  = (bf16*)take((size_t)BB * NN * CC * 2);   // 16 MB  [B][N][C]
  bf16* wqb = (bf16*)take((size_t)DD * CC * 2);
  bf16* wkb = (bf16*)take((size_t)DD * CC * 2);
  bf16* wvb = (bf16*)take((size_t)CC * CC * 2);
  bf16* wpb = (bf16*)take((size_t)CC * CC * 2);
  float* cmx = (float*)take((size_t)BB * NN * 4);
  float* csm = (float*)take((size_t)BB * NN * 4);
  float* bni = (float*)take(CC * 4);
  float* bno = (float*)take(CC * 4);

  // prep
  k_xpose<<<dim3(NN / 16, CC / 16, BB), dim3(16, 16), 0, stream>>>(x, xT);
  k_cvt<<<(DD * CC) / 256, 256, 0, stream>>>(wqb, Wq, DD * CC);
  k_cvt<<<(DD * CC) / 256, 256, 0, stream>>>(wkb, Wk, DD * CC);
  k_cvt<<<(CC * CC) / 256, 256, 0, stream>>>(wvb, Wv, CC * CC);
  k_cvt<<<(CC * CC) / 256, 256, 0, stream>>>(wpb, Wp, CC * CC);
  k_bnprep<<<1, 256, 0, stream>>>(bg, bb, bm, bv, bni, bno);

  // projections: q,k -> [B][N][64]; v -> [B][C][N] (transposed for pass-2 B-frags)
  k_proj<<<(BB * (NN / 16) * (DD / 16)) / 8, 256, 0, stream>>>(xT, wqb, qb, DD);
  k_proj<<<(BB * (NN / 16) * (DD / 16)) / 8, 256, 0, stream>>>(xT, wkb, kb, DD);
  k_projT<<<(BB * (NN / 16) * (CC / 16)) / 8, 256, 0, stream>>>(xT, wvb, vTb, CC);

  // attention
  k_colstats<<<(BB * (NN / 16)) / 8, 256, 0, stream>>>(qb, kb, cmx, csm);
  k_fsa<<<BB * (NN / 16), 128, 0, stream>>>(qb, kb, vTb, cmx, csm, x, yb);

  // LBR + residual
  k_final<<<(BB * (NN / 16) * (CC / 16)) / 8, 256, 0, stream>>>(yb, wpb, bni, bno,
                                                                x, out);
}
